// SparseAutoEncoderMNIST_62981400428626
// MI455X (gfx1250) — compile-verified, matrix operand order checked by reading, not run on hardware
//
#include <hip/hip_runtime.h>
#include <hip/hip_bf16.h>

typedef __attribute__((ext_vector_type(16))) _Float16 v16h;
typedef __attribute__((ext_vector_type(8)))  _Float16 v8h;
typedef __attribute__((ext_vector_type(8)))  float    v8f;
typedef __attribute__((ext_vector_type(4)))  int      v4i;

#if __has_builtin(__builtin_amdgcn_global_load_async_to_lds_b128)
#define SAE_HAS_ASYNC_LDS 1
#else
#define SAE_HAS_ASYNC_LDS 0
#endif

// Copy 16 bytes global -> LDS. Async (ASYNCcnt-tracked, no VGPR staging) when
// the toolchain exposes the gfx1250 async builtins; plain copy otherwise.
__device__ __forceinline__ void sae_copy16(const _Float16* g, _Float16* lds) {
#if SAE_HAS_ASYNC_LDS
  __builtin_amdgcn_global_load_async_to_lds_b128(
      (__attribute__((address_space(1))) v4i*)g,
      (__attribute__((address_space(3))) v4i*)lds, 0, 0);
#else
  *(uint4*)lds = *(const uint4*)g;
#endif
}

__device__ __forceinline__ void sae_wait_async() {
#if SAE_HAS_ASYNC_LDS
#if __has_builtin(__builtin_amdgcn_s_wait_asynccnt)
  __builtin_amdgcn_s_wait_asynccnt(0);
#else
  asm volatile("s_wait_asynccnt 0x0" ::: "memory");
#endif
#endif
}

// ---------------------------------------------------------------------------
// f32 -> f16 conversion (grid-stride)
// ---------------------------------------------------------------------------
__global__ void sae_cvt_f32_to_f16(const float* __restrict__ in,
                                   _Float16* __restrict__ out, size_t n) {
  size_t i = (size_t)blockIdx.x * blockDim.x + threadIdx.x;
  size_t stride = (size_t)gridDim.x * blockDim.x;
  for (; i < n; i += stride) out[i] = (_Float16)in[i];
}

// ---------------------------------------------------------------------------
// Column gather: y[b,k] = h[b, perm[k]]
// ---------------------------------------------------------------------------
__global__ void sae_gather_perm(const _Float16* __restrict__ h,
                                const int* __restrict__ perm,
                                _Float16* __restrict__ y) {
  const size_t total = (size_t)8192 * 256;
  size_t i = (size_t)blockIdx.x * blockDim.x + threadIdx.x;
  size_t stride = (size_t)gridDim.x * blockDim.x;
  for (; i < total; i += stride) {
    size_t b = i >> 8;
    int k = (int)(i & 255);
    y[i] = h[b * 1024 + (size_t)perm[k]];
  }
}

// ---------------------------------------------------------------------------
// WMMA GEMM: Out[m,n] = sum_k A[m,k] * W[n,k]   (A: M x K f16, W: N x K f16)
// Block tile 128x128, 8 waves (wave32), each wave 32x64 via 2x4 WMMA accs.
// Double-buffered 64-wide K tiles in LDS, async global->LDS when available.
// ---------------------------------------------------------------------------
template <int N, int K, bool OUT_F32>
__global__ __launch_bounds__(256) void sae_gemm_wmma(
    const _Float16* __restrict__ A, const _Float16* __restrict__ W,
    void* __restrict__ Out) {
  constexpr int LDA = 72;  // 64 halves + 8-half pad -> conflict-free frag reads
  __shared__ _Float16 As[2][128 * LDA];
  __shared__ _Float16 Bs[2][128 * LDA];

  const int tid = threadIdx.x;
  const int lane = tid & 31;
  const int wave = tid >> 5;
  const int wm = wave & 3;   // 4 waves along M
  const int wn = wave >> 2;  // 2 waves along N
  const int m0 = blockIdx.y * 128;
  const int n0 = blockIdx.x * 128;

  v8f acc[2][4] = {};

  const int lrow = lane & 15;
  const int kbase = (lane >> 4) * 8;  // lanes 16-31 start at K=+8 (ISA A layout)

  // stage one 128x64 tile pair into LDS buffer `buf` (4x 16B chunks/thread/mat)
  auto load_tiles = [&](int buf, int k0) {
#pragma unroll
    for (int i = 0; i < 4; ++i) {
      int c = tid + i * 256;  // 0..1023
      int row = c >> 3;       // 0..127
      int col = (c & 7) * 8;  // halves
      sae_copy16(A + (size_t)(m0 + row) * K + k0 + col,
                 &As[buf][row * LDA + col]);
      sae_copy16(W + (size_t)(n0 + row) * K + k0 + col,
                 &Bs[buf][row * LDA + col]);
    }
  };

  load_tiles(0, 0);
  sae_wait_async();
  __syncthreads();

  for (int k0 = 0; k0 < K; k0 += 64) {
    const int cur = (k0 >> 6) & 1;
    if (k0 + 64 < K) load_tiles(cur ^ 1, k0 + 64);  // prefetch next tile

#pragma unroll
    for (int kk = 0; kk < 64; kk += 32) {
      v16h afrag[2];
      v16h bfrag[4];
#pragma unroll
      for (int mf = 0; mf < 2; ++mf) {
        const _Float16* p =
            &As[cur][(wm * 32 + mf * 16 + lrow) * LDA + kk + kbase];
        v8h lo = *(const v8h*)p;
        v8h hi = *(const v8h*)(p + 16);
        afrag[mf] = __builtin_shufflevector(lo, hi, 0, 1, 2, 3, 4, 5, 6, 7, 8,
                                            9, 10, 11, 12, 13, 14, 15);
      }
#pragma unroll
      for (int nf = 0; nf < 4; ++nf) {
        const _Float16* p =
            &Bs[cur][(wn * 64 + nf * 16 + lrow) * LDA + kk + kbase];
        v8h lo = *(const v8h*)p;
        v8h hi = *(const v8h*)(p + 16);
        bfrag[nf] = __builtin_shufflevector(lo, hi, 0, 1, 2, 3, 4, 5, 6, 7, 8,
                                            9, 10, 11, 12, 13, 14, 15);
      }
#pragma unroll
      for (int mf = 0; mf < 2; ++mf)
#pragma unroll
        for (int nf = 0; nf < 4; ++nf)
          acc[mf][nf] = __builtin_amdgcn_wmma_f32_16x16x32_f16(
              false, afrag[mf], false, bfrag[nf], (short)0, acc[mf][nf], false,
              false);
    }

    sae_wait_async();  // next tile landed in LDS
    __syncthreads();   // all waves done reading `cur` / loading `cur^1`
  }

  // --- epilogue: C/D layout -> VGPR j: lanes 0-15 M=j, lanes 16-31 M=j+8
  const int mhi = (lane >> 4) * 8;
#pragma unroll
  for (int mf = 0; mf < 2; ++mf)
#pragma unroll
    for (int nf = 0; nf < 4; ++nf)
#pragma unroll
      for (int j = 0; j < 8; ++j) {
        const size_t m = (size_t)(m0 + wm * 32 + mf * 16 + mhi + j);
        const size_t n = (size_t)(n0 + wn * 64 + nf * 16 + lrow);
        const float v = acc[mf][nf][j];
        if (OUT_F32)
          ((float*)Out)[m * N + n] = v;
        else
          ((_Float16*)Out)[m * N + n] = (_Float16)v;
      }
}

// ---------------------------------------------------------------------------
// Host-side orchestration
// ---------------------------------------------------------------------------
extern "C" void kernel_launch(void* const* d_in, const int* in_sizes, int n_in,
                              void* d_out, int out_size, void* d_ws,
                              size_t ws_size, hipStream_t stream) {
  (void)in_sizes; (void)n_in; (void)out_size; (void)ws_size;
  const float* x    = (const float*)d_in[0];  // (8192,1024)
  const float* Ws   = (const float*)d_in[1];  // (10,1024,1024)
  const float* Wdec = (const float*)d_in[2];  // (1024,256)
  const int*   perm = (const int*)d_in[3];    // (256,)
  float* out = (float*)d_out;                 // (8192,1024)

  // workspace layout (f16): Wf[10M] | hA[8M] | hB[8M] | y[2M] | Wd[256K]
  _Float16* Wf = (_Float16*)d_ws;
  _Float16* hA = Wf + (size_t)10 * 1024 * 1024;
  _Float16* hB = hA + (size_t)8192 * 1024;
  _Float16* y  = hB + (size_t)8192 * 1024;
  _Float16* Wd = y + (size_t)8192 * 256;

  sae_cvt_f32_to_f16<<<4096, 256, 0, stream>>>(Ws, Wf,
                                               (size_t)10 * 1024 * 1024);
  sae_cvt_f32_to_f16<<<4096, 256, 0, stream>>>(x, hA, (size_t)8192 * 1024);
  sae_cvt_f32_to_f16<<<1024, 256, 0, stream>>>(Wdec, Wd, (size_t)1024 * 256);

  const int order[11] = {0, 1, 2, 3, 4, 5, 6, 7, 8, 8, 9};
  _Float16* src = hA;
  _Float16* dst = hB;
  const dim3 grid(1024 / 128, 8192 / 128);
  for (int i = 0; i < 11; ++i) {
    const _Float16* Wl = Wf + (size_t)order[i] * 1024 * 1024;
    sae_gemm_wmma<1024, 1024, false><<<grid, 256, 0, stream>>>(src, Wl, dst);
    _Float16* t = src; src = dst; dst = t;
  }
  // final activations are in `src`

  sae_gather_perm<<<2048, 256, 0, stream>>>(src, perm, y);

  // out = y (8192x256) @ W_dec^T  -> f32 output
  sae_gemm_wmma<1024, 256, true><<<grid, 256, 0, stream>>>(y, Wd, out);
}
